// SelfAttention_57226144252089
// MI455X (gfx1250) — matrix operand order repaired and checked by reading, not compiled, not run on hardware
//
#include <hip/hip_runtime.h>
#include <hip/hip_bf16.h>

// ---------------------------------------------------------------------------
// Sizes (fixed by the reference)
// ---------------------------------------------------------------------------
#define BATCH     2
#define SEQ       2048
#define DIM       2048
#define NHEADS    32
#define NKVHEADS  8
#define HEADDIM   64
#define NREP      4           // NHEADS / NKVHEADS
#define MROWS     (BATCH*SEQ) // 4096

typedef __attribute__((ext_vector_type(16))) __bf16 v16bf;
typedef __attribute__((ext_vector_type(8)))  float  v8f;

__device__ __forceinline__ __bf16 to_bf16(float f) { return (__bf16)f; }

// ===========================================================================
// f32 -> bf16 convert
// ===========================================================================
__global__ void cvt_f32_bf16(const float* __restrict__ in, __bf16* __restrict__ out, int n) {
    int i = blockIdx.x * blockDim.x + threadIdx.x;
    if (i < n) out[i] = to_bf16(in[i]);
}

// ===========================================================================
// WMMA GEMM: C[M,N] = A[M,K] * Bt[N,K]^T   (A,Bt bf16 row-major, C f32)
// One wave computes a 32x64 tile: 2 A-fragments x 4 B-fragments = 8 WMMAs
// per 32-wide k-step; B fragments reused across both A rows.
// block = 256 threads (8 waves) -> 256 rows x 64 cols per block.
// ===========================================================================
__global__ void gemm_bf16_wmma(const __bf16* __restrict__ A,
                               const __bf16* __restrict__ Bt,
                               float* __restrict__ C,
                               int M, int N, int K) {
    const int wave = threadIdx.x >> 5;
    const int lane = threadIdx.x & 31;
    const int hi   = lane >> 4;      // 0: lanes 0-15, 1: lanes 16-31
    const int lr   = lane & 15;

    const int m0 = blockIdx.x * 256 + wave * 32;
    const int n0 = blockIdx.y * 64;

    v8f acc[2][4];
#pragma unroll
    for (int u = 0; u < 2; ++u)
#pragma unroll
        for (int t = 0; t < 4; ++t) acc[u][t] = (v8f){};

    const __bf16* arow0 = A + (size_t)(m0 + lr) * K;
    const __bf16* arow1 = A + (size_t)(m0 + 16 + lr) * K;

    for (int k0 = 0; k0 < K; k0 += 32) {
        // ---- two A fragments: 16x32 bf16, lane = row, contiguous K (16*hi + s)
        v16bf a0, a1;
#pragma unroll
        for (int v = 0; v < 8; ++v) {
            const int koff = 2 * v + hi * 16;
            a0[2 * v]     = arow0[k0 + koff];
            a0[2 * v + 1] = arow0[k0 + koff + 1];
            a1[2 * v]     = arow1[k0 + koff];
            a1[2 * v + 1] = arow1[k0 + koff + 1];
        }
        // ---- 4 B fragments (32x16 each): lane = column, contiguous 32B ----
#pragma unroll
        for (int t = 0; t < 4; ++t) {
            const __bf16* bp = Bt + (size_t)(n0 + t * 16 + lr) * K + k0 + hi * 16;
            __builtin_prefetch(bp + 64, 0, 1);   // global_prefetch_b8
            v16bf b = *(const v16bf*)bp;
            acc[0][t] = __builtin_amdgcn_wmma_f32_16x16x32_bf16(
                false, a0, false, b, (short)0, acc[0][t], false, false);
            acc[1][t] = __builtin_amdgcn_wmma_f32_16x16x32_bf16(
                false, a1, false, b, (short)0, acc[1][t], false, false);
        }
    }

    // ---- writeback: C row = m0 + 16*u + r + 8*hi, col = n0 + 16*t + lr ----
#pragma unroll
    for (int u = 0; u < 2; ++u)
#pragma unroll
        for (int t = 0; t < 4; ++t)
#pragma unroll
            for (int r = 0; r < 8; ++r)
                C[(size_t)(m0 + u * 16 + r + hi * 8) * N + n0 + t * 16 + lr] = acc[u][t][r];
}

// ===========================================================================
// RoPE Q: qf f32 [b][s][h][64] -> qb bf16 [b][h][s][64]  (rotate pairs)
// ===========================================================================
__global__ void rope_q_kernel(const float* __restrict__ qf,
                              const float* __restrict__ cosT,
                              const float* __restrict__ sinT,
                              __bf16* __restrict__ qb) {
    int i = blockIdx.x * blockDim.x + threadIdx.x;   // BATCH*SEQ*NHEADS*32
    int p = i & 31;
    int h = (i >> 5) & 31;
    int s = (i >> 10) & 2047;
    int b = i >> 21;
    float c  = cosT[s * 32 + p];
    float sn = sinT[s * 32 + p];
    size_t src = (((size_t)b * SEQ + s) * NHEADS + h) * HEADDIM + 2 * p;
    float x0 = qf[src], x1 = qf[src + 1];
    size_t dst = (((size_t)b * NHEADS + h) * SEQ + s) * HEADDIM + 2 * p;
    qb[dst]     = to_bf16(x0 * c - x1 * sn);
    qb[dst + 1] = to_bf16(x0 * sn + x1 * c);
}

// RoPE K: kf f32 [b][s][kh][64] -> kbuf bf16 [b][kh][s][64]
__global__ void rope_k_kernel(const float* __restrict__ kf,
                              const float* __restrict__ cosT,
                              const float* __restrict__ sinT,
                              __bf16* __restrict__ kbuf) {
    int i = blockIdx.x * blockDim.x + threadIdx.x;   // BATCH*SEQ*NKVHEADS*32
    int p = i & 31;
    int h = (i >> 5) & 7;
    int s = (i >> 8) & 2047;
    int b = i >> 19;
    float c  = cosT[s * 32 + p];
    float sn = sinT[s * 32 + p];
    size_t src = (((size_t)b * SEQ + s) * NKVHEADS + h) * HEADDIM + 2 * p;
    float x0 = kf[src], x1 = kf[src + 1];
    size_t dst = (((size_t)b * NKVHEADS + h) * SEQ + s) * HEADDIM + 2 * p;
    kbuf[dst]     = to_bf16(x0 * c - x1 * sn);
    kbuf[dst + 1] = to_bf16(x0 * sn + x1 * c);
}

// V: vf f32 [b][s][kh][64] -> vbufT bf16 [b][kh][64][2048]  (TRANSPOSED)
// so the P*V B-fragment (fixed d column, contiguous keys) is one 32B load.
__global__ void conv_v_kernel(const float* __restrict__ vf, __bf16* __restrict__ vbufT) {
    int i = blockIdx.x * blockDim.x + threadIdx.x;   // BATCH*NKVHEADS*64*SEQ
    int s  = i & 2047;
    int d  = (i >> 11) & 63;
    int kh = (i >> 17) & 7;
    int b  = i >> 20;
    size_t src = (((size_t)b * SEQ + s) * NKVHEADS + kh) * HEADDIM + d;
    size_t dst = (((size_t)(b * NKVHEADS + kh) * HEADDIM + d)) * SEQ + s;
    vbufT[dst] = to_bf16(vf[src]);
}

// ===========================================================================
// Flash attention: one wave per (b, h, 16-row q tile). Online softmax,
// WMMA for QK^T (k=32 over head_dim=64 -> 2 per 16-key tile) and for P*V
// (32-key P blocks staged via LDS into the A-fragment layout).
// V is pre-transposed so V B-fragments are contiguous 32B loads.
// atb bf16 [b*s][h*64] = attention output (A of the final GEMM).
// ===========================================================================
__global__ void flash_attn_wmma(const __bf16* __restrict__ qb,
                                const __bf16* __restrict__ kb,
                                const __bf16* __restrict__ vbT,
                                __bf16* __restrict__ atb) {
    __shared__ __bf16 pl[8][16][40];   // per-wave P tile, padded stride

    const int wave = threadIdx.x >> 5;
    const int lane = threadIdx.x & 31;
    const int hi   = lane >> 4;
    const int lr   = lane & 15;

    const int gw  = blockIdx.x * 8 + wave;     // global wave id
    const int qt  = gw & 127;                  // SEQ/16
    const int h   = (gw >> 7) & 31;
    const int b   = gw >> 12;
    const int kvh = h >> 2;                    // NREP = 4
    const int q0  = qt * 16;
    const float scale = 0.125f;                // 1/sqrt(64)
    const float NEG_INF = -__builtin_inff();

    // ---- Q fragments (held in registers for the whole pass) ----
    const __bf16* qrow = qb + (((size_t)(b * NHEADS + h)) * SEQ + q0 + lr) * HEADDIM;
    v16bf qa[2];
#pragma unroll
    for (int c = 0; c < 2; ++c)
#pragma unroll
        for (int v = 0; v < 8; ++v) {
            const int koff = c * 32 + 2 * v + hi * 16;
            qa[c][2 * v]     = qrow[koff];
            qa[c][2 * v + 1] = qrow[koff + 1];
        }

    const __bf16* kbase = kb  + ((size_t)(b * NKVHEADS + kvh)) * SEQ * HEADDIM;
    const __bf16* vbase = vbT + ((size_t)(b * NKVHEADS + kvh)) * HEADDIM * SEQ;

    float mrow[8], lrow[8];
#pragma unroll
    for (int r = 0; r < 8; ++r) { mrow[r] = NEG_INF; lrow[r] = 0.0f; }
    v8f o[4];
    o[0] = (v8f){}; o[1] = (v8f){}; o[2] = (v8f){}; o[3] = (v8f){};

    for (int kb0 = 0; kb0 <= q0 + 15; kb0 += 32) {
        // ---- scores for two 16-key subtiles ----
        v8f sc[2];
#pragma unroll
        for (int sub = 0; sub < 2; ++sub) {
            const int k0 = kb0 + sub * 16;
            const __bf16* krow = kbase + (size_t)(k0 + lr) * HEADDIM + hi * 16;
            v16bf kf0 = *(const v16bf*)(krow);        // d = 0..31 chunk
            v16bf kf1 = *(const v16bf*)(krow + 32);   // d = 32..63 chunk
            v8f s = (v8f){};
            s = __builtin_amdgcn_wmma_f32_16x16x32_bf16(false, qa[0], false, kf0,
                                                        (short)0, s, false, false);
            s = __builtin_amdgcn_wmma_f32_16x16x32_bf16(false, qa[1], false, kf1,
                                                        (short)0, s, false, false);
            sc[sub] = s;
        }
        // ---- scale + causal mask ----
#pragma unroll
        for (int sub = 0; sub < 2; ++sub)
#pragma unroll
            for (int r = 0; r < 8; ++r) {
                const int col = kb0 + sub * 16 + lr;
                const int row = q0 + r + hi * 8;
                float v = sc[sub][r] * scale;
                sc[sub][r] = (col > row) ? NEG_INF : v;
            }
        // ---- online softmax update (row reductions inside 16-lane groups) ----
#pragma unroll
        for (int r = 0; r < 8; ++r) {
            float bm = fmaxf(sc[0][r], sc[1][r]);
            bm = fmaxf(bm, __shfl_xor(bm, 1));
            bm = fmaxf(bm, __shfl_xor(bm, 2));
            bm = fmaxf(bm, __shfl_xor(bm, 4));
            bm = fmaxf(bm, __shfl_xor(bm, 8));
            const float mnew = fmaxf(mrow[r], bm);
            const float f    = __expf(mrow[r] - mnew);
            const float p0   = __expf(sc[0][r] - mnew);
            const float p1   = __expf(sc[1][r] - mnew);
            float ps = p0 + p1;
            ps += __shfl_xor(ps, 1);
            ps += __shfl_xor(ps, 2);
            ps += __shfl_xor(ps, 4);
            ps += __shfl_xor(ps, 8);
            lrow[r] = lrow[r] * f + ps;
            mrow[r] = mnew;
#pragma unroll
            for (int t = 0; t < 4; ++t) o[t][r] *= f;
            pl[wave][r + hi * 8][lr]      = to_bf16(p0);
            pl[wave][r + hi * 8][16 + lr] = to_bf16(p1);
        }
        // ---- reload P (16x32) from LDS in A-fragment layout ----
        v16bf pa;
#pragma unroll
        for (int v = 0; v < 8; ++v) {
            const int koff = 2 * v + hi * 16;
            pa[2 * v]     = pl[wave][lr][koff];
            pa[2 * v + 1] = pl[wave][lr][koff + 1];
        }
        // ---- P * V : 4 WMMAs (d chunks of 16), contiguous loads from V^T ----
#pragma unroll
        for (int t = 0; t < 4; ++t) {
            v16bf vfr = *(const v16bf*)(vbase + (size_t)(t * 16 + lr) * SEQ + kb0 + hi * 16);
            o[t] = __builtin_amdgcn_wmma_f32_16x16x32_bf16(false, pa, false, vfr,
                                                           (short)0, o[t], false, false);
        }
    }

    // ---- normalize and write bf16 output [b*s][h*64] ----
#pragma unroll
    for (int r = 0; r < 8; ++r) {
        const float inv = 1.0f / lrow[r];
        const size_t row = (size_t)b * SEQ + q0 + r + hi * 8;
#pragma unroll
        for (int t = 0; t < 4; ++t)
            atb[row * DIM + h * HEADDIM + t * 16 + lr] = to_bf16(o[t][r] * inv);
    }
}

// ===========================================================================
// Host launcher
// ===========================================================================
extern "C" void kernel_launch(void* const* d_in, const int* in_sizes, int n_in,
                              void* d_out, int out_size, void* d_ws, size_t ws_size,
                              hipStream_t stream) {
    const float* x  = (const float*)d_in[0];
    const float* fc = (const float*)d_in[1];
    const float* fs = (const float*)d_in[2];
    const float* wq = (const float*)d_in[3];
    const float* wk = (const float*)d_in[4];
    const float* wv = (const float*)d_in[5];
    const float* wo = (const float*)d_in[6];
    float* out = (float*)d_out;
    char* ws = (char*)d_ws;

    // workspace layout (256B aligned offsets)
    const size_t OFF_XB   = 0;                              // 4096x2048 bf16
    const size_t OFF_WQB  = OFF_XB   + (size_t)MROWS * DIM * 2;        // 16.8MB
    const size_t OFF_WKB  = OFF_WQB  + (size_t)DIM * DIM * 2;          // +8.4MB
    const size_t OFF_WVB  = OFF_WKB  + (size_t)(NKVHEADS*HEADDIM) * DIM * 2;
    const size_t OFF_WOB  = OFF_WVB  + (size_t)(NKVHEADS*HEADDIM) * DIM * 2;
    const size_t OFF_QB   = OFF_WOB  + (size_t)DIM * DIM * 2;
    const size_t OFF_KB   = OFF_QB   + (size_t)BATCH*NHEADS*SEQ*HEADDIM * 2;
    const size_t OFF_VB   = OFF_KB   + (size_t)BATCH*NKVHEADS*SEQ*HEADDIM * 2;
    const size_t OFF_KF   = OFF_VB   + (size_t)BATCH*NKVHEADS*SEQ*HEADDIM * 2;
    const size_t OFF_VF   = OFF_KF   + (size_t)MROWS * (NKVHEADS*HEADDIM) * 4;
    const size_t OFF_QF   = OFF_VF   + (size_t)MROWS * (NKVHEADS*HEADDIM) * 4;
    const size_t OFF_ATB  = OFF_QF;  // atb (bf16) aliases qf (f32, dead after RoPE)

    __bf16* xb   = (__bf16*)(ws + OFF_XB);
    __bf16* wqb  = (__bf16*)(ws + OFF_WQB);
    __bf16* wkb  = (__bf16*)(ws + OFF_WKB);
    __bf16* wvb  = (__bf16*)(ws + OFF_WVB);
    __bf16* wob  = (__bf16*)(ws + OFF_WOB);
    __bf16* qbuf = (__bf16*)(ws + OFF_QB);
    __bf16* kbuf = (__bf16*)(ws + OFF_KB);
    __bf16* vbufT= (__bf16*)(ws + OFF_VB);
    float*  kf   = (float*) (ws + OFF_KF);
    float*  vf   = (float*) (ws + OFF_VF);
    float*  qf   = (float*) (ws + OFF_QF);
    __bf16* atb  = (__bf16*)(ws + OFF_ATB);

    // 1) f32 -> bf16 conversions
    {
        int n;
        n = MROWS * DIM;          cvt_f32_bf16<<<(n+255)/256, 256, 0, stream>>>(x,  xb,  n);
        n = DIM * DIM;            cvt_f32_bf16<<<(n+255)/256, 256, 0, stream>>>(wq, wqb, n);
        n = NKVHEADS*HEADDIM*DIM; cvt_f32_bf16<<<(n+255)/256, 256, 0, stream>>>(wk, wkb, n);
        n = NKVHEADS*HEADDIM*DIM; cvt_f32_bf16<<<(n+255)/256, 256, 0, stream>>>(wv, wvb, n);
        n = DIM * DIM;            cvt_f32_bf16<<<(n+255)/256, 256, 0, stream>>>(wo, wob, n);
    }

    // 2) QKV projections (WMMA bf16, f32 out); 256 rows x 64 cols per block
    gemm_bf16_wmma<<<dim3(MROWS/256, DIM/64),               256, 0, stream>>>(xb, wqb, qf, MROWS, DIM,               DIM);
    gemm_bf16_wmma<<<dim3(MROWS/256, (NKVHEADS*HEADDIM)/64),256, 0, stream>>>(xb, wkb, kf, MROWS, NKVHEADS*HEADDIM, DIM);
    gemm_bf16_wmma<<<dim3(MROWS/256, (NKVHEADS*HEADDIM)/64),256, 0, stream>>>(xb, wvb, vf, MROWS, NKVHEADS*HEADDIM, DIM);

    // 3) RoPE + relayout to bf16 (V goes to a transposed [d][s] layout)
    {
        int n;
        n = BATCH*SEQ*NHEADS*32;     rope_q_kernel<<<(n+255)/256, 256, 0, stream>>>(qf, fc, fs, qbuf);
        n = BATCH*SEQ*NKVHEADS*32;   rope_k_kernel<<<(n+255)/256, 256, 0, stream>>>(kf, fc, fs, kbuf);
        n = BATCH*NKVHEADS*HEADDIM*SEQ; conv_v_kernel<<<(n+255)/256, 256, 0, stream>>>(vf, vbufT);
    }

    // 4) causal flash attention (one wave per 16 q rows)
    {
        int nwaves = BATCH * NHEADS * (SEQ / 16);   // 8192
        flash_attn_wmma<<<nwaves / 8, 256, 0, stream>>>(qbuf, kbuf, vbufT, atb);
    }

    // 5) output projection -> d_out (f32)
    gemm_bf16_wmma<<<dim3(MROWS/256, DIM/64), 256, 0, stream>>>(atb, wob, out, MROWS, DIM, DIM);
}
